// GINAtomBondClassifier_82076825027187
// MI455X (gfx1250) — matrix operand order repaired and checked by reading, not compiled, not run on hardware
//
#include <hip/hip_runtime.h>

#define NN 60000
#define NE 180000
#define NG 2048
#define HD 256
#define HH (HD*HD)

typedef __attribute__((ext_vector_type(16))) __bf16 v16bf;
typedef __attribute__((ext_vector_type(8)))  float  v8f;

struct U4 { unsigned int x, y, z, w; };
struct F4 { float x, y, z, w; };

union BFrag {
    v16bf v;
    __bf16 e[16];
    U4 q[2];
};

// ---------------------------------------------------------------------------
// Generic 256-wide GEMM:  Out[M,256] = epilogue(A[M,256] @ Wt^T + ...)
// Wt is stored transposed+bf16: Wt[n*256 + k].
// Block = 256 threads = 8 waves; block covers 32 rows x 256 cols.
// Wave (w&1) selects 16-row half; (w>>1) selects 64-col quarter (4 N-tiles).
// AMODE: 0 = fp32 A; 1 = bf16 A; 2 = fp32 A scaled by rowscale[m] (mean-pool)
// EMODE: 0 = relu(acc+bias) -> bf16 out
//        1 = relu(acc*scale[n] + shift[n]) -> f32 out   (bias folded in shift)
//        2 = outf[m,n] += acc + bias[n]                  (VN residual update)
//        3 = relu(acc+bias) -> f32 out
// ---------------------------------------------------------------------------
template <int AMODE, int EMODE>
__global__ __launch_bounds__(256) void gemm256_kernel(
    const void* __restrict__ Av, const __bf16* __restrict__ Wt,
    const float* __restrict__ bias, const float* __restrict__ bnsc,
    const float* __restrict__ bnsh, const float* __restrict__ rowscale,
    float* __restrict__ outf, __bf16* __restrict__ outb)
{
    const int lane  = threadIdx.x & 31;
    const int wave  = threadIdx.x >> 5;
    const int mBase = blockIdx.x * 32 + (wave & 1) * 16;
    const int nBase = (wave >> 1) * 64;
    const int ml    = lane & 15;
    const int half  = lane >> 4;
    const int m     = mBase + ml;

    v8f zero = {0.f, 0.f, 0.f, 0.f, 0.f, 0.f, 0.f, 0.f};
    v8f acc[4] = {zero, zero, zero, zero};

    float rs = 1.0f;
    if (AMODE == 2) rs = rowscale[m];

    for (int kc = 0; kc < 8; ++kc) {
        BFrag a;
        if (AMODE == 1) {
            const __bf16* Ap = (const __bf16*)Av + (size_t)m * 256 + kc * 32 + half * 8;
            a.q[0] = *(const U4*)(Ap);        // K: base+0..7
            a.q[1] = *(const U4*)(Ap + 16);   // K: base+16..23
        } else {
            const float* Ap = (const float*)Av + (size_t)m * 256 + kc * 32 + half * 8;
            F4 f0 = *(const F4*)(Ap);
            F4 f1 = *(const F4*)(Ap + 4);
            F4 f2 = *(const F4*)(Ap + 16);
            F4 f3 = *(const F4*)(Ap + 20);
            a.e[0]  = (__bf16)(f0.x * rs); a.e[1]  = (__bf16)(f0.y * rs);
            a.e[2]  = (__bf16)(f0.z * rs); a.e[3]  = (__bf16)(f0.w * rs);
            a.e[4]  = (__bf16)(f1.x * rs); a.e[5]  = (__bf16)(f1.y * rs);
            a.e[6]  = (__bf16)(f1.z * rs); a.e[7]  = (__bf16)(f1.w * rs);
            a.e[8]  = (__bf16)(f2.x * rs); a.e[9]  = (__bf16)(f2.y * rs);
            a.e[10] = (__bf16)(f2.z * rs); a.e[11] = (__bf16)(f2.w * rs);
            a.e[12] = (__bf16)(f3.x * rs); a.e[13] = (__bf16)(f3.y * rs);
            a.e[14] = (__bf16)(f3.z * rs); a.e[15] = (__bf16)(f3.w * rs);
        }
#pragma unroll
        for (int t = 0; t < 4; ++t) {
            BFrag b;
            const __bf16* Bp = Wt + (size_t)(nBase + t * 16 + ml) * 256 + kc * 32 + half * 8;
            b.q[0] = *(const U4*)(Bp);
            b.q[1] = *(const U4*)(Bp + 16);
            acc[t] = __builtin_amdgcn_wmma_f32_16x16x32_bf16(
                false, a.v, false, b.v, (short)0, acc[t], false, false);
        }
    }

    const int mrow0 = mBase + half * 8;
#pragma unroll
    for (int t = 0; t < 4; ++t) {
        const int n = nBase + t * 16 + ml;
        float sA = 0.f, sB = 0.f, bv = 0.f;
        if (EMODE == 1) { sA = bnsc[n]; sB = bnsh[n]; }
        else            { bv = bias[n]; }
#pragma unroll
        for (int r = 0; r < 8; ++r) {
            float v = acc[t][r];
            size_t idx = (size_t)(mrow0 + r) * 256 + n;
            if (EMODE == 0) {
                v += bv; v = v > 0.f ? v : 0.f;
                outb[idx] = (__bf16)v;
            } else if (EMODE == 1) {
                v = v * sA + sB; v = v > 0.f ? v : 0.f;
                outf[idx] = v;
            } else if (EMODE == 2) {
                outf[idx] = outf[idx] + v + bv;
            } else {
                v += bv; v = v > 0.f ? v : 0.f;
                outf[idx] = v;
            }
        }
    }
}

// --------------------------- support kernels -------------------------------

__global__ void convT_kernel(const float* __restrict__ src, __bf16* __restrict__ dst)
{
    int i = blockIdx.x * 256 + threadIdx.x;   // 0..65535
    int n = i >> 8, k = i & 255;
    dst[(size_t)n * 256 + k] = (__bf16)src[(size_t)k * 256 + n];
}

__global__ void bn_prep_kernel(const float* __restrict__ gamma, const float* __restrict__ beta,
                               const float* __restrict__ mean, const float* __restrict__ var,
                               const float* __restrict__ b2, float* __restrict__ sc,
                               float* __restrict__ sh)
{
    int i = blockIdx.x * 256 + threadIdx.x;   // layer = blockIdx.x
    float s = gamma[i] * rsqrtf(var[i] + 1e-5f);
    sc[i] = s;
    sh[i] = (b2[i] - mean[i]) * s + beta[i];
}

__global__ void zero_kernel(float* p, int n)
{
    int i = blockIdx.x * 256 + threadIdx.x;
    if (i < n) p[i] = 0.f;
}

__global__ void count_kernel(const int* __restrict__ batch, float* __restrict__ cnt)
{
    int i = blockIdx.x * 256 + threadIdx.x;
    if (i < NN) atomicAdd(&cnt[batch[i]], 1.0f);
}

__global__ void invcnt_kernel(const float* __restrict__ cnt, float* __restrict__ inv)
{
    int i = blockIdx.x * 256 + threadIdx.x;
    inv[i] = 1.0f / fmaxf(cnt[i], 1.0f);
}

__global__ void vn_init_kernel(float* __restrict__ vn, const float* __restrict__ vn_emb)
{
    int i = blockIdx.x * 256 + threadIdx.x;
    vn[i] = vn_emb[i & 255];
}

__global__ void node_embed_kernel(const int* __restrict__ x, const float* __restrict__ emb,
                                  float* __restrict__ h)
{
    int i = blockIdx.x * 256 + threadIdx.x;
    int n = i >> 8, c = i & 255;
    const int offs[9] = {0, 119, 124, 136, 148, 158, 164, 170, 172};
    const int maxs[9] = {118, 4, 11, 11, 9, 5, 5, 1, 1};
    float s = 0.f;
#pragma unroll
    for (int j = 0; j < 9; ++j) {
        int v = x[n * 9 + j];
        v = v < 0 ? 0 : (v > maxs[j] ? maxs[j] : v);
        s += emb[(size_t)(v + offs[j]) * 256 + c];
    }
    h[i] = s;
}

// h += vn[batch];  zacc = (1+eps_l) * h   (fused)
__global__ void addvn_zinit_kernel(float* __restrict__ h, float* __restrict__ zacc,
                                   const float* __restrict__ vn, const int* __restrict__ batch,
                                   const float* __restrict__ eps, int l)
{
    int i = blockIdx.x * 256 + threadIdx.x;
    int n = i >> 8, c = i & 255;
    float v = h[i] + vn[(size_t)batch[n] * 256 + c];
    h[i] = v;
    zacc[i] = (1.0f + eps[l]) * v;
}

// per edge: msg = relu(h[src] + edge_emb_sum); zacc[dst] += msg (atomic)
__global__ void edge_scatter_kernel(const float* __restrict__ h, const int* __restrict__ ei,
                                    const int* __restrict__ ea, const float* __restrict__ emb,
                                    float* __restrict__ zacc)
{
    int tid = blockIdx.x * 256 + threadIdx.x;
    int e = tid >> 6;                 // 64 threads per edge
    int c = (tid & 63) << 2;          // 4 channels per thread
    int src = ei[e];
    int dst = ei[NE + e];
    int a0 = ea[e * 3 + 0]; a0 = a0 < 0 ? 0 : (a0 > 4 ? 4 : a0);
    int a1 = ea[e * 3 + 1]; a1 = a1 < 0 ? 0 : (a1 > 5 ? 5 : a1); a1 += 5;
    int a2 = ea[e * 3 + 2]; a2 = a2 < 0 ? 0 : (a2 > 1 ? 1 : a2); a2 += 11;
    F4 hv = *(const F4*)(h + (size_t)src * 256 + c);
    F4 e0 = *(const F4*)(emb + (size_t)a0 * 256 + c);
    F4 e1 = *(const F4*)(emb + (size_t)a1 * 256 + c);
    F4 e2 = *(const F4*)(emb + (size_t)a2 * 256 + c);
    float m0 = hv.x + e0.x + e1.x + e2.x; m0 = m0 > 0.f ? m0 : 0.f;
    float m1 = hv.y + e0.y + e1.y + e2.y; m1 = m1 > 0.f ? m1 : 0.f;
    float m2 = hv.z + e0.z + e1.z + e2.z; m2 = m2 > 0.f ? m2 : 0.f;
    float m3 = hv.w + e0.w + e1.w + e2.w; m3 = m3 > 0.f ? m3 : 0.f;
    float* p = zacc + (size_t)dst * 256 + c;
    atomicAdd(p + 0, m0);
    atomicAdd(p + 1, m1);
    atomicAdd(p + 2, m2);
    atomicAdd(p + 3, m3);
}

__global__ void pool_kernel(const float* __restrict__ h, const int* __restrict__ batch,
                            float* __restrict__ gsum)
{
    int i = blockIdx.x * 256 + threadIdx.x;
    int n = i >> 8, c = i & 255;
    atomicAdd(&gsum[(size_t)batch[n] * 256 + c], h[i]);
}

// out[g] = dot(hhid[g,:], hW2) + hb2   (hhid already relu'd)
__global__ void head_dot_kernel(const float* __restrict__ hhid, const float* __restrict__ hW2,
                                const float* __restrict__ hb2, float* __restrict__ out)
{
    int wave = threadIdx.x >> 5;
    int lane = threadIdx.x & 31;
    int g = blockIdx.x * 8 + wave;
    const float* row = hhid + (size_t)g * 256;
    float s = 0.f;
#pragma unroll
    for (int i = 0; i < 8; ++i) s += row[lane + i * 32] * hW2[lane + i * 32];
#pragma unroll
    for (int off = 16; off > 0; off >>= 1) s += __shfl_xor(s, off, 32);
    if (lane == 0) out[g] = s + hb2[0];
}

// ---------------------------------------------------------------------------

extern "C" void kernel_launch(void* const* d_in, const int* in_sizes, int n_in,
                              void* d_out, int out_size, void* d_ws, size_t ws_size,
                              hipStream_t stream)
{
    const int*   x         = (const int*)d_in[0];
    const int*   edge_idx  = (const int*)d_in[1];
    const int*   edge_attr = (const int*)d_in[2];
    const int*   batch     = (const int*)d_in[3];
    const float* node_emb  = (const float*)d_in[4];
    const float* edge_emb  = (const float*)d_in[5];
    const float* vn_emb    = (const float*)d_in[6];
    const float* eps       = (const float*)d_in[7];
    const float* W1        = (const float*)d_in[8];
    const float* b1        = (const float*)d_in[9];
    const float* W2        = (const float*)d_in[10];
    const float* b2        = (const float*)d_in[11];
    const float* bn_gamma  = (const float*)d_in[12];
    const float* bn_beta   = (const float*)d_in[13];
    const float* bn_mean   = (const float*)d_in[14];
    const float* bn_var    = (const float*)d_in[15];
    const float* vW1       = (const float*)d_in[16];
    const float* vb1       = (const float*)d_in[17];
    const float* vW2       = (const float*)d_in[18];
    const float* vb2       = (const float*)d_in[19];
    const float* hW1       = (const float*)d_in[20];
    const float* hb1       = (const float*)d_in[21];
    const float* hW2       = (const float*)d_in[22];
    const float* hb2       = (const float*)d_in[23];

    char* ws = (char*)d_ws;
    size_t off = 0;
    auto alloc = [&](size_t bytes) -> void* {
        void* p = ws + off;
        off += (bytes + 255) & ~(size_t)255;
        return p;
    };

    float*  h     = (float*)alloc((size_t)NN * HD * 4);
    float*  zacc  = (float*)alloc((size_t)NN * HD * 4);
    __bf16* ybf   = (__bf16*)alloc((size_t)NN * HD * 2);
    float*  vn    = (float*)alloc((size_t)NG * HD * 4);
    float*  gsum  = (float*)alloc((size_t)NG * HD * 4);
    __bf16* vybf  = (__bf16*)alloc((size_t)NG * HD * 2);
    float*  hhid  = (float*)alloc((size_t)NG * HD * 4);
    float*  cnt   = (float*)alloc((size_t)NG * 4);
    float*  invc  = (float*)alloc((size_t)NG * 4);
    __bf16* WtAll = (__bf16*)alloc((size_t)21 * HH * 2);
    float*  bnsc  = (float*)alloc((size_t)5 * HD * 4);
    float*  bnsh  = (float*)alloc((size_t)5 * HD * 4);

    // ---- prep: weight transpose+bf16 convert (all 21 256x256 matrices) ----
    for (int l = 0; l < 5; ++l) {
        convT_kernel<<<256, 256, 0, stream>>>(W1 + (size_t)l * HH, WtAll + (size_t)l * HH);
        convT_kernel<<<256, 256, 0, stream>>>(W2 + (size_t)l * HH, WtAll + (size_t)(5 + l) * HH);
        convT_kernel<<<256, 256, 0, stream>>>(vW1 + (size_t)l * HH, WtAll + (size_t)(10 + l) * HH);
        convT_kernel<<<256, 256, 0, stream>>>(vW2 + (size_t)l * HH, WtAll + (size_t)(15 + l) * HH);
    }
    convT_kernel<<<256, 256, 0, stream>>>(hW1, WtAll + (size_t)20 * HH);

    bn_prep_kernel<<<5, 256, 0, stream>>>(bn_gamma, bn_beta, bn_mean, bn_var, b2, bnsc, bnsh);

    zero_kernel<<<(NG + 255) / 256, 256, 0, stream>>>(cnt, NG);
    count_kernel<<<(NN + 255) / 256, 256, 0, stream>>>(batch, cnt);
    invcnt_kernel<<<(NG + 255) / 256, 256, 0, stream>>>(cnt, invc);

    vn_init_kernel<<<NG * HD / 256, 256, 0, stream>>>(vn, vn_emb);
    node_embed_kernel<<<NN, 256, 0, stream>>>(x, node_emb, h);

    const int MB = NN / 32;   // 1875 gemm blocks for node rows
    const int GB = NG / 32;   // 64 gemm blocks for graph rows

    for (int l = 0; l < 5; ++l) {
        addvn_zinit_kernel<<<NN, 256, 0, stream>>>(h, zacc, vn, batch, eps, l);
        edge_scatter_kernel<<<NE * 64 / 256, 256, 0, stream>>>(h, edge_idx, edge_attr, edge_emb, zacc);

        // z @ W1 + b1 -> relu -> ybf (bf16)
        gemm256_kernel<0, 0><<<MB, 256, 0, stream>>>(
            zacc, WtAll + (size_t)l * HH, b1 + (size_t)l * HD,
            nullptr, nullptr, nullptr, nullptr, ybf);
        // ybf @ W2 (+b2, BN folded) -> relu -> h (f32)
        gemm256_kernel<1, 1><<<MB, 256, 0, stream>>>(
            ybf, WtAll + (size_t)(5 + l) * HH, nullptr,
            bnsc + (size_t)l * HD, bnsh + (size_t)l * HD, nullptr, h, nullptr);

        // mean pool
        zero_kernel<<<NG * HD / 256, 256, 0, stream>>>(gsum, NG * HD);
        pool_kernel<<<NN, 256, 0, stream>>>(h, batch, gsum);

        // VN MLP: relu(g @ vW1 + vb1) @ vW2 + vb2, vn += result
        gemm256_kernel<2, 0><<<GB, 256, 0, stream>>>(
            gsum, WtAll + (size_t)(10 + l) * HH, vb1 + (size_t)l * HD,
            nullptr, nullptr, invc, nullptr, vybf);
        gemm256_kernel<1, 2><<<GB, 256, 0, stream>>>(
            vybf, WtAll + (size_t)(15 + l) * HH, vb2 + (size_t)l * HD,
            nullptr, nullptr, nullptr, vn, nullptr);
    }

    // final pooling + head
    zero_kernel<<<NG * HD / 256, 256, 0, stream>>>(gsum, NG * HD);
    pool_kernel<<<NN, 256, 0, stream>>>(h, batch, gsum);
    gemm256_kernel<2, 3><<<GB, 256, 0, stream>>>(
        gsum, WtAll + (size_t)20 * HH, hb1, nullptr, nullptr, invc, hhid, nullptr);
    head_dot_kernel<<<NG / 8, 256, 0, stream>>>(hhid, hW2, hb2, (float*)d_out);
}
